// FusedExpertsNetwork_31155692765926
// MI455X (gfx1250) — compile-verified, matrix operand order checked by reading, not run on hardware
//
#include <hip/hip_runtime.h>
#include <hip/hip_bf16.h>

// Problem dims (match reference)
constexpr int NE = 8;     // experts
constexpr int NC = 1024;  // capacity (tokens per expert)
constexpr int NM = 2048;  // model dim
constexpr int NH = 8192;  // hidden dim
constexpr int NO = 2048;  // output dim

typedef __attribute__((ext_vector_type(16))) __bf16 v16bf;
typedef __attribute__((ext_vector_type(8)))  __bf16 v8bf;
typedef __attribute__((ext_vector_type(4)))  __bf16 v4bf;
typedef __attribute__((ext_vector_type(8)))  float  v8f;
typedef __attribute__((ext_vector_type(4)))  float  v4f;

// LDS tile row stride in bf16 elements: 32 data + 8 pad (16B) -> conflict-free b128 reads
constexpr int LDSK = 40;

union Frag { v16bf v; v8bf h[2]; };

__device__ __forceinline__ void wait_async0() {
  asm volatile("s_wait_asynccnt 0" ::: "memory");
}

// Per-lane async copy of 16 bytes global -> LDS (ASYNCcnt-tracked, no VGPR round trip)
__device__ __forceinline__ void async_copy_b128(const void* gptr, const void* lptr) {
  // Flat LDS addresses carry the wave-relative LDS byte offset in the low 32 bits.
  uint32_t loff = (uint32_t)(uintptr_t)lptr;
  asm volatile("global_load_async_to_lds_b128 %0, %1, off"
               :: "v"(loff), "v"(gptr) : "memory");
}

// ---------------------------------------------------------------------------
// FC1: y[e,c,h] = relu( sum_m x[e,c,m] * w1[e,h,m] + b1[e,h] ), stored bf16
// Double-buffered LDS, 2-stage pipeline: global->reg prefetch of tile t+1
// overlaps the WMMA block on tile t. One barrier per K-step.
// ---------------------------------------------------------------------------
__global__ __launch_bounds__(256) void fc1_kernel(const float* __restrict__ x,
                                                  const float* __restrict__ w1,
                                                  const float* __restrict__ b1,
                                                  __bf16* __restrict__ y) {
  __shared__ alignas(16) __bf16 As[2][128 * LDSK];
  __shared__ alignas(16) __bf16 Bs[2][128 * LDSK];

  const int e  = blockIdx.z;
  const int c0 = blockIdx.x * 128;
  const int h0 = blockIdx.y * 128;

  const int tid   = threadIdx.x;
  const int lane  = tid & 31;
  const int wid   = tid >> 5;
  const int wm    = wid & 3;   // 4 waves along C, 32 rows each
  const int wn    = wid >> 2;  // 2 waves along H, 64 cols each
  const int mlane = lane & 15;
  const int kb    = (lane >> 4) * 8;  // ISA bf16 fragment K-split per half-wave

  // Staging slot for this thread (constant across K steps): 4 chunks of f32x4
  const int srow = tid >> 3;        // base row, +32 per chunk
  const int scv  = (tid & 7) * 4;   // 0..28

  const float* xa = x  + ((size_t)e * NC + c0) * NM;
  const float* wb = w1 + ((size_t)e * NH + h0) * NM;

  v8f acc[2][4];
#pragma unroll
  for (int i = 0; i < 2; ++i)
#pragma unroll
    for (int j = 0; j < 4; ++j) acc[i][j] = (v8f)0.0f;

  v4f ra[4], rb[4];

  auto load_tiles = [&](int k0) {
#pragma unroll
    for (int p = 0; p < 4; ++p) {
      const size_t off = (size_t)(srow + p * 32) * NM + k0 + scv;
      ra[p] = *(const v4f*)(xa + off);
      rb[p] = *(const v4f*)(wb + off);
    }
  };
  auto store_tiles = [&](int buf) {
#pragma unroll
    for (int p = 0; p < 4; ++p) {
      const int off = (srow + p * 32) * LDSK + scv;
      v4bf ap, bp;
#pragma unroll
      for (int q = 0; q < 4; ++q) { ap[q] = (__bf16)ra[p][q]; bp[q] = (__bf16)rb[p][q]; }
      *(v4bf*)&As[buf][off] = ap;
      *(v4bf*)&Bs[buf][off] = bp;
    }
  };
  auto compute = [&](int buf) {
    Frag af[2], bf[4];
#pragma unroll
    for (int i = 0; i < 2; ++i) {
      const int row = wm * 32 + i * 16 + mlane;
      af[i].h[0] = *(const v8bf*)&As[buf][row * LDSK + kb];
      af[i].h[1] = *(const v8bf*)&As[buf][row * LDSK + 16 + kb];
    }
#pragma unroll
    for (int j = 0; j < 4; ++j) {
      const int row = wn * 64 + j * 16 + mlane;
      bf[j].h[0] = *(const v8bf*)&Bs[buf][row * LDSK + kb];
      bf[j].h[1] = *(const v8bf*)&Bs[buf][row * LDSK + 16 + kb];
    }
#pragma unroll
    for (int i = 0; i < 2; ++i)
#pragma unroll
      for (int j = 0; j < 4; ++j)
        acc[i][j] = __builtin_amdgcn_wmma_f32_16x16x32_bf16(
            false, af[i].v, false, bf[j].v, (short)0, acc[i][j], false, false);
  };

  constexpr int NT = NM / 32;  // 64 K-steps
  load_tiles(0);
  store_tiles(0);
  __syncthreads();

  for (int t = 0; t < NT; ++t) {
    const int cur = t & 1;
    if (t + 1 < NT) load_tiles((t + 1) * 32);  // global prefetch overlaps WMMA
    compute(cur);
    if (t + 1 < NT) store_tiles(cur ^ 1);
    __syncthreads();
  }

  // Epilogue: bias + ReLU, store bf16 activations
#pragma unroll
  for (int j = 0; j < 4; ++j) {
    const int h    = h0 + wn * 64 + j * 16 + mlane;
    const float bv = b1[e * NH + h];
#pragma unroll
    for (int i = 0; i < 2; ++i) {
      const int mbase = c0 + wm * 32 + i * 16 + (lane >> 4) * 8;
#pragma unroll
      for (int r = 0; r < 8; ++r) {
        float v = acc[i][j][r] + bv;
        v = v > 0.0f ? v : 0.0f;
        y[((size_t)e * NC + (mbase + r)) * NH + h] = (__bf16)v;
      }
    }
  }
}

// ---------------------------------------------------------------------------
// FC2: out[e,c,o] = sum_h y[e,c,h] * w2[e,h,o] + b2[e,o]   (f32 output)
// A tile is already bf16 -> async global->LDS copy (ASYNCcnt path).
// B tile (f32, K-strided) converts + transposes through registers.
// Double-buffered LDS, one barrier per K-step.
// ---------------------------------------------------------------------------
__global__ __launch_bounds__(256) void fc2_kernel(const __bf16* __restrict__ y,
                                                  const float* __restrict__ w2,
                                                  const float* __restrict__ b2,
                                                  float* __restrict__ out) {
  __shared__ alignas(16) __bf16 As[2][128 * LDSK];
  __shared__ alignas(16) __bf16 Bs[2][128 * LDSK];

  const int e  = blockIdx.z;
  const int c0 = blockIdx.x * 128;
  const int o0 = blockIdx.y * 128;

  const int tid   = threadIdx.x;
  const int lane  = tid & 31;
  const int wid   = tid >> 5;
  const int wm    = wid & 3;
  const int wn    = wid >> 2;
  const int mlane = lane & 15;
  const int kb    = (lane >> 4) * 8;

  // A staging: 2 async b128 chunks per thread
  const int arow = tid >> 2;        // base row, +64 per chunk
  const int acv  = (tid & 3) * 8;   // bf16 element offset, 16B granular
  // B staging: 4 f32x4 chunks per thread, transposed on LDS store
  const int bk   = tid >> 5;        // base k, +8 per chunk  (q>>5 with q=tid+p*256)
  const int bn4  = (tid & 31) * 4;  // 0..124

  const __bf16* ya = y  + ((size_t)e * NC + c0) * NH;
  const float*  wp = w2 + (size_t)e * NH * NO;

  v8f acc[2][4];
#pragma unroll
  for (int i = 0; i < 2; ++i)
#pragma unroll
    for (int j = 0; j < 4; ++j) acc[i][j] = (v8f)0.0f;

  v4f rb[4];

  auto issueA = [&](int k0, int buf) {
#pragma unroll
    for (int p = 0; p < 2; ++p) {
      const int row = arow + p * 64;
      async_copy_b128(ya + (size_t)row * NH + k0 + acv,
                      &As[buf][row * LDSK + acv]);
    }
  };
  auto loadB = [&](int k0) {
#pragma unroll
    for (int p = 0; p < 4; ++p)
      rb[p] = *(const v4f*)(wp + (size_t)(k0 + bk + p * 8) * NO + o0 + bn4);
  };
  auto storeB = [&](int buf) {
#pragma unroll
    for (int p = 0; p < 4; ++p) {
      const int k = bk + p * 8;
#pragma unroll
      for (int s = 0; s < 4; ++s)
        Bs[buf][(bn4 + s) * LDSK + k] = (__bf16)rb[p][s];
    }
  };
  auto compute = [&](int buf) {
    Frag af[2], bf[4];
#pragma unroll
    for (int i = 0; i < 2; ++i) {
      const int row = wm * 32 + i * 16 + mlane;
      af[i].h[0] = *(const v8bf*)&As[buf][row * LDSK + kb];
      af[i].h[1] = *(const v8bf*)&As[buf][row * LDSK + 16 + kb];
    }
#pragma unroll
    for (int j = 0; j < 4; ++j) {
      const int row = wn * 64 + j * 16 + mlane;
      bf[j].h[0] = *(const v8bf*)&Bs[buf][row * LDSK + kb];
      bf[j].h[1] = *(const v8bf*)&Bs[buf][row * LDSK + 16 + kb];
    }
#pragma unroll
    for (int i = 0; i < 2; ++i)
#pragma unroll
      for (int j = 0; j < 4; ++j)
        acc[i][j] = __builtin_amdgcn_wmma_f32_16x16x32_bf16(
            false, af[i].v, false, bf[j].v, (short)0, acc[i][j], false, false);
  };

  constexpr int NT = NH / 32;  // 256 K-steps
  issueA(0, 0);
  loadB(0);
  storeB(0);
  wait_async0();
  __syncthreads();

  for (int t = 0; t < NT; ++t) {
    const int cur = t & 1;
    if (t + 1 < NT) {
      issueA((t + 1) * 32, cur ^ 1);  // async copy runs during WMMA below
      loadB((t + 1) * 32);
    }
    compute(cur);
    if (t + 1 < NT) {
      storeB(cur ^ 1);
      wait_async0();                  // A-tile copy visible before barrier
    }
    __syncthreads();
  }

  // Epilogue: bias add, f32 store (lane-contiguous along O -> coalesced)
#pragma unroll
  for (int j = 0; j < 4; ++j) {
    const int o    = o0 + wn * 64 + j * 16 + mlane;
    const float bv = b2[e * NO + o];
#pragma unroll
    for (int i = 0; i < 2; ++i) {
      const int mbase = c0 + wm * 32 + i * 16 + (lane >> 4) * 8;
#pragma unroll
      for (int r = 0; r < 8; ++r)
        out[((size_t)e * NC + (mbase + r)) * NO + o] = acc[i][j][r] + bv;
    }
  }
}

// ---------------------------------------------------------------------------
extern "C" void kernel_launch(void* const* d_in, const int* in_sizes, int n_in,
                              void* d_out, int out_size, void* d_ws, size_t ws_size,
                              hipStream_t stream) {
  const float* x  = (const float*)d_in[0];   // [E,C,M]
  const float* w1 = (const float*)d_in[1];   // [E,H,M]
  const float* b1 = (const float*)d_in[2];   // [E,H]
  const float* w2 = (const float*)d_in[3];   // [E,H,O]
  const float* b2 = (const float*)d_in[4];   // [E,O]
  float* out = (float*)d_out;                // [E,C,O]

  __bf16* ymid = (__bf16*)d_ws;              // [E,C,H] bf16 activations (134 MB)

  dim3 blk(256);
  dim3 g1(NC / 128, NH / 128, NE);           // 8 x 64 x 8 blocks
  fc1_kernel<<<g1, blk, 0, stream>>>(x, w1, b1, ymid);

  dim3 g2(NC / 128, NO / 128, NE);           // 8 x 16 x 8 blocks
  fc2_kernel<<<g2, blk, 0, stream>>>(ymid, w2, b2, out);
}
